// LoRA_MoElayer_87479893885604
// MI455X (gfx1250) — matrix operand order repaired and checked by reading, not compiled, not run on hardware
//
#include <hip/hip_runtime.h>
#include <hip/hip_bf16.h>

typedef __attribute__((ext_vector_type(16))) __bf16 v16bf;
typedef __attribute__((ext_vector_type(8)))  float  v8f;

#define N_TOK     32768
#define DIM       1024
#define NEXP      7
#define KAPPA_INV 10.0f
#define SINK_EPS  1e-6f
#define NORM_EPS  1e-12f
#define LOG_EPSF  2.220446049250313e-16f

// ---------------- workspace layout (bytes) ----------------
#define L_OFF     0                           // float [7][N]    sinkhorn matrix
#define PN_OFF    (L_OFF + NEXP*N_TOK*4)      // float [7][1024] normalized prototypes
#define PART_OFF  (PN_OFF + NEXP*DIM*4)       // float [128][8]  block-partial row sums
#define RS_OFF    (PART_OFF + 128*8*4)        // float [8]       row sums
#define CNT_OFF   (RS_OFF + 64)               // int   [8]       per-expert counts
#define OFF_OFF   (CNT_OFF + 64)              // int   [8]       per-expert offsets
#define EIDX_OFF  (OFF_OFF + 64)              // int   [N]       chosen expert
#define POS_OFF   (EIDX_OFF + N_TOK*4)        // int   [N]       slot within expert
#define LOGG_OFF  (POS_OFF + N_TOK*4)         // float [N]       log(gate)
#define TLIST_OFF (LOGG_OFF + N_TOK*4)        // int   [N]       tokens grouped by expert
#define ASWZ_OFF  (TLIST_OFF + N_TOK*4)       // bf16 swizzled A weights (fragment-major)
// B swizzle offset computed at launch (after all A blocks)

// ---------------- routing kernels ----------------
__global__ void k_zero(int* __restrict__ cnt) {
  if (threadIdx.x < 8) cnt[threadIdx.x] = 0;
}

__global__ void k_pnorm(const float* __restrict__ pr, float* __restrict__ pn) {
  __shared__ float red[256];
  const int k = blockIdx.x, tid = threadIdx.x;
  float s = 0.f;
  for (int d = tid; d < DIM; d += 256) { float v = pr[k*DIM + d]; s += v*v; }
  red[tid] = s; __syncthreads();
  for (int o = 128; o > 0; o >>= 1) { if (tid < o) red[tid] += red[tid+o]; __syncthreads(); }
  const float inv = 1.0f / fmaxf(sqrtf(red[0]), NORM_EPS);
  for (int d = tid; d < DIM; d += 256) pn[k*DIM + d] = pr[k*DIM + d] * inv;
}

__global__ void k_sim(const float* __restrict__ x, const float* __restrict__ pn,
                      float* __restrict__ L, float* __restrict__ part) {
  __shared__ float spn[NEXP*DIM];
  __shared__ float red[256];
  const int tid = threadIdx.x;
  for (int i = tid; i < NEXP*DIM; i += 256) spn[i] = pn[i];
  __syncthreads();
  const int n = blockIdx.x * 256 + tid;            // grid covers N exactly
  float dot[NEXP] = {0,0,0,0,0,0,0};
  float xx = 0.f;
  const float4* xr = (const float4*)(x + (size_t)n * DIM);
  for (int d4 = 0; d4 < DIM/4; ++d4) {
    float4 v = xr[d4];
    xx += v.x*v.x + v.y*v.y + v.z*v.z + v.w*v.w;
    #pragma unroll
    for (int k = 0; k < NEXP; ++k) {
      const float* p = spn + k*DIM + d4*4;
      dot[k] += v.x*p[0] + v.y*p[1] + v.z*p[2] + v.w*p[3];
    }
  }
  const float inv = 1.0f / fmaxf(sqrtf(xx), NORM_EPS);
  float pk[NEXP];
  #pragma unroll
  for (int k = 0; k < NEXP; ++k) {
    float l = expf(dot[k] * inv * KAPPA_INV);
    L[k*N_TOK + n] = l;
    pk[k] = l;
  }
  #pragma unroll
  for (int k = 0; k < NEXP; ++k) {                 // deterministic block partials
    red[tid] = pk[k]; __syncthreads();
    for (int o = 128; o > 0; o >>= 1) { if (tid < o) red[tid] += red[tid+o]; __syncthreads(); }
    if (tid == 0) part[blockIdx.x*8 + k] = red[0];
    __syncthreads();
  }
}

__global__ void k_reduce(const float* __restrict__ part, float* __restrict__ rs, int nb) {
  const int k = threadIdx.x;
  if (k < NEXP) {
    float s = 0.f;
    for (int b = 0; b < nb; ++b) s += part[b*8 + k];   // fixed order: deterministic
    rs[k] = s;
  }
}

__global__ void k_sink_mid(float* __restrict__ L, const float* __restrict__ rs,
                           float* __restrict__ part) {
  __shared__ float sinv[NEXP];
  __shared__ float red[256];
  const int tid = threadIdx.x;
  if (tid < NEXP) sinv[tid] = 1.0f / (rs[tid] + SINK_EPS);
  __syncthreads();
  const int n = blockIdx.x * 256 + tid;
  float v[NEXP]; float cs = 0.f;
  #pragma unroll
  for (int k = 0; k < NEXP; ++k) { v[k] = L[k*N_TOK + n] * sinv[k]; cs += v[k]; }
  const float ic = 1.0f / (cs + SINK_EPS);
  float pk[NEXP];
  #pragma unroll
  for (int k = 0; k < NEXP; ++k) { float w = v[k]*ic; L[k*N_TOK + n] = w; pk[k] = w; }
  #pragma unroll
  for (int k = 0; k < NEXP; ++k) {
    red[tid] = pk[k]; __syncthreads();
    for (int o = 128; o > 0; o >>= 1) { if (tid < o) red[tid] += red[tid+o]; __syncthreads(); }
    if (tid == 0) part[blockIdx.x*8 + k] = red[0];
    __syncthreads();
  }
}

__global__ void k_sink_last(const float* __restrict__ L, const float* __restrict__ rs,
                            float* __restrict__ logg, int* __restrict__ eidx,
                            int* __restrict__ pos, int* __restrict__ cnt) {
  __shared__ float sinv[NEXP];
  const int tid = threadIdx.x;
  if (tid < NEXP) sinv[tid] = 1.0f / (rs[tid] + SINK_EPS);
  __syncthreads();
  const int n = blockIdx.x * 256 + tid;
  float v[NEXP]; float cs = 0.f;
  #pragma unroll
  for (int k = 0; k < NEXP; ++k) { v[k] = L[k*N_TOK + n] * sinv[k]; cs += v[k]; }
  const float ic = 1.0f / (cs + SINK_EPS);
  float best = -1.f; int be = 0;
  #pragma unroll
  for (int k = 0; k < NEXP; ++k) { float w = v[k]*ic; if (w > best) { best = w; be = k; } }
  logg[n] = logf(fmaxf(best, LOG_EPSF));
  eidx[n] = be;
  pos[n]  = atomicAdd(cnt + be, 1);   // int atomic: counts deterministic
}

__global__ void k_offsets(const int* __restrict__ cnt, int* __restrict__ offs) {
  if (threadIdx.x == 0) { int o = 0; for (int e = 0; e < NEXP; ++e) { offs[e] = o; o += cnt[e]; } }
}

__global__ void k_scatter(const int* __restrict__ eidx, const int* __restrict__ pos,
                          const int* __restrict__ offs, int* __restrict__ tlist) {
  const int n = blockIdx.x * 256 + threadIdx.x;
  const int e = eidx[n];
  tlist[offs[e] + pos[n]] = n;
}

// ---------------- weight pre-swizzle: f32 -> bf16 in WMMA B-fragment order ----
// A_swz frag f = kbi*NACC + jc; element addr ((f*32 + lane)*16 + i):
//   value = Ae[jc*16 + nn][kbi*32 + 16*h + i]   (0 if row >= D)
// B_swz frag g = c0i*NK2 + kk:
//   value = Be[c0i*16 + nn][kk*32 + 16*h + i]   (0 if k >= D)
template<int D>
__global__ void k_swz(const float* __restrict__ Ae, const float* __restrict__ Be,
                      __bf16* __restrict__ Aswz, __bf16* __restrict__ Bswz) {
  constexpr int NACC = (D + 15) / 16;
  constexpr int NK2  = (D + 31) / 32;
  const int nA = (DIM/32) * NACC * 512;   // elems
  const int nB = (DIM/16) * NK2  * 512;
  const int stride = gridDim.x * 256;
  for (int t = blockIdx.x*256 + threadIdx.x; t < nA; t += stride) {
    int i = t & 15, lane = (t >> 4) & 31, f = t >> 9;
    int jc = f % NACC, kbi = f / NACC;
    int h = lane >> 4, nn = lane & 15;
    int k = kbi*32 + 16*h + i;
    int rowi = jc*16 + nn;
    float val = (rowi < D) ? Ae[(size_t)rowi*DIM + k] : 0.f;
    Aswz[t] = (__bf16)val;
  }
  for (int t = blockIdx.x*256 + threadIdx.x; t < nB; t += stride) {
    int i = t & 15, lane = (t >> 4) & 31, g = t >> 9;
    int kk = g % NK2, c0i = g / NK2;
    int h = lane >> 4, nn = lane & 15;
    int k = kk*32 + 16*h + i;
    float val = (k < D) ? Be[(size_t)(c0i*16 + nn)*D + k] : 0.f;
    Bswz[t] = (__bf16)val;
  }
}

// ---------------- expert LoRA pass (WMMA bf16, f32 accumulate) ----------------
template<int D>
__global__ __launch_bounds__(128) void k_expert(
    const float* __restrict__ x,
    const v16bf* __restrict__ Aswz, const v16bf* __restrict__ Bswz,
    const int* __restrict__ tlist, const float* __restrict__ logg,
    const int* __restrict__ cnt,   const int* __restrict__ offs,
    float* __restrict__ out, int eid)
{
  constexpr int NACC = (D + 15) / 16;   // U column chunks (GEMM1 accumulators)
  constexpr int NK2  = (D + 31) / 32;   // GEMM2 K steps
  constexpr int KPAD = NK2 * 32;        // zero-padded K for GEMM2
  constexpr int LS   = KPAD + 8;        // LDS row stride (bf16 units)

  __shared__ __bf16 Uall[4][16 * LS];
  __shared__ int    sTok[4][16];
  __shared__ float  sLg [4][16];

  const int wave = threadIdx.x >> 5;
  const int lane = threadIdx.x & 31;
  const int h    = lane >> 4;           // half-wave
  const int nn   = lane & 15;

  const int count = cnt[eid];
  const int base  = offs[eid];
  const int tile  = blockIdx.x * 4 + wave;
  const int tbase = tile * 16;
  if (tbase >= count) return;           // wave-uniform exit; no barriers used below
  const int rem = count - tbase;        // >= 1

  __bf16* U = &Uall[wave][0];

  if (h == 0) {
    int idx = (nn < rem) ? (base + tbase + nn) : base;   // clamp tail to a valid token
    int t = tlist[idx];
    sTok[wave][nn] = t;
    sLg [wave][nn] = logg[t];
  }
  if constexpr (KPAD > NACC * 16) {     // zero pad columns of U once
    constexpr int PW = KPAD - NACC * 16;
    for (int i = lane; i < 16 * PW; i += 32) {
      int r = i / PW, c = NACC*16 + (i % PW);
      U[r * LS + c] = (__bf16)0.0f;
    }
  }
  __threadfence_block();                // LDS is in-order within a wave

  const int    myTok = sTok[wave][nn];
  const float* xrow  = x + (size_t)myTok * DIM;

  float lgS[8]; int tokS[8]; bool valS[8];
  #pragma unroll
  for (int j = 0; j < 8; ++j) {
    int m = j + 8*h;
    lgS[j]  = sLg [wave][m];
    tokS[j] = sTok[wave][m];
    valS[j] = (m < rem);
  }

  // ---- GEMM1: U = Xg(16x1024) * Ae^T ----
  v8f acc[NACC] = {};
  #pragma unroll 2
  for (int kbi = 0; kbi < DIM/32; ++kbi) {
    v16bf a;                                          // A-frag: row = token (gathered)
    #pragma unroll
    for (int v = 0; v < 8; ++v) {
      int k = kbi*32 + 16*h + 2*v;
      float2 p = *(const float2*)(xrow + k);
      a[2*v] = (__bf16)p.x; a[2*v+1] = (__bf16)p.y;
    }
    const v16bf* af = Aswz + (size_t)kbi*NACC*32 + lane;
    #pragma unroll
    for (int jc = 0; jc < NACC; ++jc) {
      v16bf b = af[jc*32];                            // one 32B coalesced load, pre-swizzled
      acc[jc] = __builtin_amdgcn_wmma_f32_16x16x32_bf16(false, a, false, b,
                                                        (short)0, acc[jc], false, false);
    }
  }

  // ---- stage U (D-layout -> row-major bf16 in LDS) ----
  #pragma unroll
  for (int jc = 0; jc < NACC; ++jc)
    #pragma unroll
    for (int j = 0; j < 8; ++j) {
      int m = j + 8*h;                                 // token row
      U[m * LS + jc*16 + nn] = (__bf16)acc[jc][j];
    }
  __threadfence_block();

  // ---- reload U as A-fragments ----
  v16bf uf[NK2];
  #pragma unroll
  for (int kk = 0; kk < NK2; ++kk) {
    #pragma unroll
    for (int v = 0; v < 8; ++v) {
      int k = kk*32 + 16*h + 2*v;
      uf[kk][2*v]   = U[nn * LS + k];
      uf[kk][2*v+1] = U[nn * LS + k + 1];
    }
  }

  // ---- GEMM2 + epilogue ----
  for (int c0i = 0; c0i < DIM/16; ++c0i) {
    v8f y = {};
    const v16bf* bf = Bswz + (size_t)c0i*NK2*32 + lane;
    #pragma unroll
    for (int kk = 0; kk < NK2; ++kk) {
      v16bf b = bf[kk*32];                            // one 32B coalesced load, pre-swizzled
      y = __builtin_amdgcn_wmma_f32_16x16x32_bf16(false, uf[kk], false, b,
                                                  (short)0, y, false, false);
    }
    #pragma unroll
    for (int j = 0; j < 8; ++j)
      if (valS[j]) out[(size_t)tokS[j] * DIM + c0i*16 + nn] = y[j] + lgS[j];
  }
}

// ---------------- launcher ----------------
extern "C" void kernel_launch(void* const* d_in, const int* in_sizes, int n_in,
                              void* d_out, int out_size, void* d_ws, size_t ws_size,
                              hipStream_t stream) {
  (void)in_sizes; (void)n_in; (void)out_size; (void)ws_size;
  const float* x  = (const float*)d_in[0];
  const float* pr = (const float*)d_in[1];
  const float* A[NEXP]; const float* B[NEXP];
  for (int e = 0; e < NEXP; ++e) { A[e] = (const float*)d_in[2 + 2*e]; B[e] = (const float*)d_in[3 + 2*e]; }
  float* out = (float*)d_out;
  char*  w   = (char*)d_ws;
  float* L    = (float*)(w + L_OFF);
  float* pn   = (float*)(w + PN_OFF);
  float* part = (float*)(w + PART_OFF);
  float* rs   = (float*)(w + RS_OFF);
  int*   cnt  = (int*)  (w + CNT_OFF);
  int*   offs = (int*)  (w + OFF_OFF);
  int*   eidx = (int*)  (w + EIDX_OFF);
  int*   pos  = (int*)  (w + POS_OFF);
  float* logg = (float*)(w + LOGG_OFF);
  int*   tl   = (int*)  (w + TLIST_OFF);

  // per-expert swizzled-weight offsets (bytes)
  static const int NACCe[NEXP] = {1,1,2,3,4,6,8};
  static const int NK2e [NEXP] = {1,1,1,2,2,3,4};
  __bf16* Aswz[NEXP]; __bf16* Bswz[NEXP];
  size_t o = ASWZ_OFF;
  for (int e = 0; e < NEXP; ++e) { Aswz[e] = (__bf16*)(w + o); o += (size_t)(DIM/32)*NACCe[e]*512*2; }
  for (int e = 0; e < NEXP; ++e) { Bswz[e] = (__bf16*)(w + o); o += (size_t)(DIM/16)*NK2e [e]*512*2; }

  // weight pre-swizzle (tiny; L2-resident afterwards)
  k_swz<  8><<<64, 256, 0, stream>>>(A[0], B[0], Aswz[0], Bswz[0]);
  k_swz< 16><<<64, 256, 0, stream>>>(A[1], B[1], Aswz[1], Bswz[1]);
  k_swz< 32><<<64, 256, 0, stream>>>(A[2], B[2], Aswz[2], Bswz[2]);
  k_swz< 48><<<64, 256, 0, stream>>>(A[3], B[3], Aswz[3], Bswz[3]);
  k_swz< 64><<<64, 256, 0, stream>>>(A[4], B[4], Aswz[4], Bswz[4]);
  k_swz< 96><<<64, 256, 0, stream>>>(A[5], B[5], Aswz[5], Bswz[5]);
  k_swz<128><<<64, 256, 0, stream>>>(A[6], B[6], Aswz[6], Bswz[6]);

  // routing
  k_zero     <<<1,    64, 0, stream>>>(cnt);
  k_pnorm    <<<NEXP, 256, 0, stream>>>(pr, pn);
  k_sim      <<<128,  256, 0, stream>>>(x, pn, L, part);
  k_reduce   <<<1,    32, 0, stream>>>(part, rs, 128);
  k_sink_mid <<<128,  256, 0, stream>>>(L, rs, part);
  k_reduce   <<<1,    32, 0, stream>>>(part, rs, 128);
  k_sink_mid <<<128,  256, 0, stream>>>(L, rs, part);
  k_reduce   <<<1,    32, 0, stream>>>(part, rs, 128);
  k_sink_last<<<128,  256, 0, stream>>>(L, rs, logg, eidx, pos, cnt);
  k_offsets  <<<1,    32, 0, stream>>>(cnt, offs);
  k_scatter  <<<128,  256, 0, stream>>>(eidx, pos, offs, tl);

  // expert pass
  dim3 g(512), blk(128);  // 512 blocks * 4 waves = 2048 tiles = worst case N/16
  k_expert<  8><<<g, blk, 0, stream>>>(x, (const v16bf*)Aswz[0], (const v16bf*)Bswz[0], tl, logg, cnt, offs, out, 0);
  k_expert< 16><<<g, blk, 0, stream>>>(x, (const v16bf*)Aswz[1], (const v16bf*)Bswz[1], tl, logg, cnt, offs, out, 1);
  k_expert< 32><<<g, blk, 0, stream>>>(x, (const v16bf*)Aswz[2], (const v16bf*)Bswz[2], tl, logg, cnt, offs, out, 2);
  k_expert< 48><<<g, blk, 0, stream>>>(x, (const v16bf*)Aswz[3], (const v16bf*)Bswz[3], tl, logg, cnt, offs, out, 3);
  k_expert< 64><<<g, blk, 0, stream>>>(x, (const v16bf*)Aswz[4], (const v16bf*)Bswz[4], tl, logg, cnt, offs, out, 4);
  k_expert< 96><<<g, blk, 0, stream>>>(x, (const v16bf*)Aswz[5], (const v16bf*)Bswz[5], tl, logg, cnt, offs, out, 5);
  k_expert<128><<<g, blk, 0, stream>>>(x, (const v16bf*)Aswz[6], (const v16bf*)Bswz[6], tl, logg, cnt, offs, out, 6);
}
